// MoEBitNetFFN_12249246729067
// MI455X (gfx1250) — 8-bit matrix operand order repaired and checked by reading, compiled, not run on hardware
//
#include <hip/hip_runtime.h>
#include <cstdint>


#define N_TOK   8192
#define DM      1024
#define DF      4096
#define NEXP    16
#define CAP     768
#define NROWS   (NEXP*CAP)

typedef int v8i __attribute__((ext_vector_type(8)));
typedef unsigned int v4u __attribute__((ext_vector_type(4)));
typedef int v4s __attribute__((ext_vector_type(4)));

#if __has_builtin(__builtin_amdgcn_tensor_load_to_lds)
#define HAVE_TDM 1
#else
#define HAVE_TDM 0
#endif


static __device__ __forceinline__ float gelu_tanh(float v) {
    const float c = 0.7978845608028654f;
    float t = tanhf(c * (v + 0.044715f * v * v * v));
    return 0.5f * v * (1.0f + t);
}

static __device__ __forceinline__ int q8(float v, float s) {
    float q = rintf(v * s);
    q = fminf(fmaxf(q, -128.0f), 127.0f);
    return ((int)q) & 0xff;
}

static __device__ __forceinline__ uint32_t pack4(float4 v, float s) {
    return (uint32_t)q8(v.x, s) | ((uint32_t)q8(v.y, s) << 8) |
           ((uint32_t)q8(v.z, s) << 16) | ((uint32_t)q8(v.w, s) << 24);
}

static __device__ __forceinline__ float wave_max(float m) {
    #pragma unroll
    for (int off = 16; off; off >>= 1) m = fmaxf(m, __shfl_xor(m, off, 32));
    return m;
}

static __device__ __forceinline__ v8i load_a_frag(const char* a, int lane) {
    int row  = lane & 15;
    int ksel = (lane >> 4) << 4;
    const char* p = a + row * 80 + ksel;
    v8i r;
    #pragma unroll
    for (int pi = 0; pi < 4; ++pi) {
        uint64_t v = *(const uint64_t*)(p + (pi >> 1) * 32 + (pi & 1) * 8);
        r[2 * pi]     = (int)(uint32_t)v;
        r[2 * pi + 1] = (int)(uint32_t)(v >> 32);
    }
    return r;
}

static __device__ __forceinline__ v8i load_b_frag(const char* bt, int lane) {
    int n  = lane & 15;
    int kh = (lane >> 4) << 4;
    const char* p = bt + n * 80 + kh;
    uint4 q0 = *(const uint4*)(p);
    uint4 q1 = *(const uint4*)(p + 32);
    v8i r;
    r[0] = q0.x; r[1] = q0.y; r[2] = q0.z; r[3] = q0.w;
    r[4] = q1.x; r[5] = q1.y; r[6] = q1.z; r[7] = q1.w;
    return r;
}

#if HAVE_TDM
static __device__ __forceinline__ void tdm_load_tile(const void* gp, uint32_t lds_off,
                                                     uint32_t gstride) {
    uint64_t ga = (uint64_t)(uintptr_t)gp;
    v4u g0;
    g0[0] = 1u;
    g0[1] = lds_off;
    g0[2] = (uint32_t)ga;
    g0[3] = ((uint32_t)(ga >> 32) & 0x01FFFFFFu) | (2u << 30);
    v8i g1;
    g1[0] = (int)((1u << 20) | (3u << 22) | (3u << 25));
    g1[1] = (int)((gstride & 0xFFFFu) << 16);
    g1[2] = (int)(((gstride >> 16) & 0xFFFFu) | (128u << 16));
    g1[3] = (int)(64u << 16);
    g1[4] = (int)(128u);
    g1[5] = (int)gstride;
    g1[6] = 0;
    g1[7] = 0;
    v4s gz4 = {0, 0, 0, 0};
    v8i gz8 = {0, 0, 0, 0, 0, 0, 0, 0};
    __builtin_amdgcn_tensor_load_to_lds(g0, g1, gz4, gz4, gz8, 0);
}
#endif


__global__ void k_zero_small(float* wsum, float* fcnt, float* psum) {
    int t = threadIdx.x;
    if (t < 32) wsum[t] = 0.0f;
    if (t < 16) { fcnt[t] = 0.0f; psum[t] = 0.0f; }
}

__global__ void k_zero_f32(float* p, int n) {
    int i = blockIdx.x * 256 + threadIdx.x;
    if (i < n) p[i] = 0.0f;
}

__global__ void k_zero_u32(uint32_t* p, int n) {
    int i = blockIdx.x * 256 + threadIdx.x;
    if (i < n) p[i] = 0u;
}

__global__ void k_init_ascale(float* p, int n) {
    int i = blockIdx.x * 256 + threadIdx.x;
    if (i < n) p[i] = 1e-5f / 127.0f;
}

__global__ void k_wabs_sum(const float* __restrict__ w, float* __restrict__ sums) {
    __shared__ float red[256];
    int e = blockIdx.y;
    size_t base = (size_t)e * ((size_t)DM * DF) + (size_t)blockIdx.x * 16384 + threadIdx.x;
    float acc = 0.0f;
    #pragma unroll 8
    for (int j = 0; j < 64; ++j) acc += fabsf(w[base + (size_t)j * 256]);
    red[threadIdx.x] = acc;
    __syncthreads();
    for (int s = 128; s; s >>= 1) {
        if (threadIdx.x < s) red[threadIdx.x] += red[threadIdx.x + s];
        __syncthreads();
    }
    if (threadIdx.x == 0) atomicAdd(&sums[e], red[0]);
}

__global__ void k_wscale(const float* sums, float* scales, float inv) {
    int i = threadIdx.x;
    if (i < 32) scales[i] = fmaxf(sums[i] * inv, 1e-5f);
}

__global__ void k_quant_w_t(const float* __restrict__ w, const float* __restrict__ scale,
                            int8_t* __restrict__ q, int R, int C) {
    size_t idx = (size_t)blockIdx.x * 256 + threadIdx.x;
    size_t total = (size_t)NEXP * R * C;
    if (idx >= total) return;
    size_t rc = (size_t)R * C;
    int e = (int)(idx / rc);
    size_t rem = idx - (size_t)e * rc;
    int r = (int)(rem / C);
    int c = (int)(rem - (size_t)r * C);
    float v = rintf(w[idx] / scale[e]);
    v = fminf(fmaxf(v, -1.0f), 1.0f);
    q[((size_t)e * C + c) * R + r] = (int8_t)v;
}


__global__ void k_router(const float* __restrict__ x, const float* __restrict__ rw,
                         int* __restrict__ tokE, float* __restrict__ tokW,
                         float* __restrict__ fcnt, float* __restrict__ psum) {
    int lane = threadIdx.x & 31;
    int wave = threadIdx.x >> 5;
    int n = blockIdx.x * 8 + wave;
    const float* xr = x + (size_t)n * DM;
    float p[NEXP];
    #pragma unroll
    for (int e = 0; e < NEXP; ++e) p[e] = 0.0f;
    for (int j = 0; j < DM / 32; ++j) {
        int d = lane + j * 32;
        float xv = xr[d];
        const float* rwd = rw + (size_t)d * NEXP;
        #pragma unroll
        for (int e = 0; e < NEXP; ++e) p[e] += xv * rwd[e];
    }
    #pragma unroll
    for (int off = 16; off; off >>= 1) {
        #pragma unroll
        for (int e = 0; e < NEXP; ++e) p[e] += __shfl_xor(p[e], off, 32);
    }
    if (lane == 0) {
        float mx = p[0];
        #pragma unroll
        for (int e = 1; e < NEXP; ++e) mx = fmaxf(mx, p[e]);
        float ex[NEXP], sum = 0.0f;
        #pragma unroll
        for (int e = 0; e < NEXP; ++e) { ex[e] = __expf(p[e] - mx); sum += ex[e]; }
        float inv = 1.0f / sum;
        int b1 = 0; float v1 = -1.0f;
        #pragma unroll
        for (int e = 0; e < NEXP; ++e) { float pr = ex[e] * inv; if (pr > v1) { v1 = pr; b1 = e; } }
        int b2 = (b1 == 0) ? 1 : 0; float v2 = -1.0f;
        #pragma unroll
        for (int e = 0; e < NEXP; ++e) {
            if (e == b1) continue;
            float pr = ex[e] * inv; if (pr > v2) { v2 = pr; b2 = e; }
        }
        float denom = v1 + v2 + 1e-8f;
        tokE[n]         = b1; tokW[n]         = v1 / denom;
        tokE[N_TOK + n] = b2; tokW[N_TOK + n] = v2 / denom;
        atomicAdd(&fcnt[b1], 1.0f);
        atomicAdd(&fcnt[b2], 1.0f);
        #pragma unroll
        for (int e = 0; e < NEXP; ++e) atomicAdd(&psum[e], ex[e] * inv);
    }
}


__global__ void k_count_chunks(const int* __restrict__ te, int* __restrict__ cc) {
    __shared__ int h[NEXP];
    int t = threadIdx.x;
    if (t < NEXP) h[t] = 0;
    __syncthreads();
    atomicAdd(&h[te[blockIdx.x * 256 + t]], 1);
    __syncthreads();
    if (t < NEXP) cc[blockIdx.x * NEXP + t] = h[t];
}

__global__ void k_scan_chunks(const int* __restrict__ cc, int* __restrict__ cb) {
    int e = threadIdx.x;
    if (e < NEXP) {
        int base = 0;
        for (int c = 0; c < N_TOK / 256; ++c) {
            cb[c * NEXP + e] = base;
            base += cc[c * NEXP + e];
        }
    }
}

__global__ void k_rank_local(const int* __restrict__ te, const int* __restrict__ cb,
                             int* __restrict__ tl) {
    __shared__ int eid[256];
    int t = threadIdx.x;
    int n = blockIdx.x * 256 + t;
    int e = te[n];
    eid[t] = e;
    __syncthreads();
    int r = 0;
    for (int j = 0; j < t; ++j) r += (eid[j] == e);
    int l = cb[blockIdx.x * NEXP + e] + r;
    tl[n] = (l > CAP - 1) ? (CAP - 1) : l;
}

__global__ void k_quant_scatter(const float* __restrict__ x, const int* __restrict__ te,
                                const int* __restrict__ tl, int8_t* __restrict__ bufq,
                                float* __restrict__ ascale) {
    int lane = threadIdx.x & 31;
    int wave = threadIdx.x >> 5;
    int n = blockIdx.x * 8 + wave;
    int e = te[n], l = tl[n];
    const float4* xr = (const float4*)(x + (size_t)n * DM);
    float m = 0.0f;
    #pragma unroll
    for (int j = 0; j < DM / 128; ++j) {
        float4 v = xr[lane + j * 32];
        m = fmaxf(m, fmaxf(fmaxf(fabsf(v.x), fabsf(v.y)), fmaxf(fabsf(v.z), fabsf(v.w))));
    }
    m = wave_max(m);
    float mx = fmaxf(m, 1e-5f);
    float s = 127.0f / mx;
    uint32_t* row = (uint32_t*)(bufq + ((size_t)e * CAP + l) * DM);
    #pragma unroll
    for (int j = 0; j < DM / 128; ++j) {
        int i = lane + j * 32;
        row[i] = pack4(xr[i], s);
    }
    if (lane == 0) ascale[e * CAP + l] = mx / 127.0f;
}

__global__ void k_row_quant(const float* __restrict__ in, int cols,
                            int8_t* __restrict__ q, float* __restrict__ scale) {
    int lane = threadIdx.x & 31;
    int wave = threadIdx.x >> 5;
    int row = blockIdx.x * 8 + wave;
    const float4* r = (const float4*)(in + (size_t)row * cols);
    float m = 0.0f;
    for (int i = lane; i < cols / 4; i += 32) {
        float4 v = r[i];
        m = fmaxf(m, fmaxf(fmaxf(fabsf(v.x), fabsf(v.y)), fmaxf(fabsf(v.z), fabsf(v.w))));
    }
    m = wave_max(m);
    float mx = fmaxf(m, 1e-5f);
    float s = 127.0f / mx;
    uint32_t* qr = (uint32_t*)(q + (size_t)row * cols);
    for (int i = lane; i < cols / 4; i += 32) qr[i] = pack4(r[i], s);
    if (lane == 0) scale[row] = mx / 127.0f;
}


template <bool GELU>
__global__ __launch_bounds__(256) void k_moe_gemm_i8(
        const int8_t* __restrict__ Aq, const float* __restrict__ aScale,
        const int8_t* __restrict__ BTq, const float* __restrict__ wScale,
        float* __restrict__ Out, int NN, int K) {
    __shared__ __align__(16) char As[2][128 * 80];
    __shared__ __align__(16) char Bs[2][128 * 80];

    const int e = blockIdx.z;
    const int nBase = blockIdx.x * 128;
    const int mBase = blockIdx.y * 128;
    const int t = threadIdx.x;
    const int lane = t & 31;
    const int wave = t >> 5;
    const int wm = wave & 3;
    const int wn = wave >> 2;

    const int8_t* gA = Aq  + ((size_t)e * CAP + mBase) * K;
    const int8_t* gB = BTq + ((size_t)e * NN  + nBase) * K;

#if !HAVE_TDM
    const int c0 = t, c1 = t + 256;
    const int r0 = c0 >> 2, o0 = (c0 & 3) << 4;
    const int r1 = c1 >> 2, o1 = (c1 & 3) << 4;
#endif

    v8i acc[2][4];
    const v8i vzero = {0, 0, 0, 0, 0, 0, 0, 0};
    #pragma unroll
    for (int mi = 0; mi < 2; ++mi)
        #pragma unroll
        for (int ni = 0; ni < 4; ++ni) acc[mi][ni] = vzero;

#if HAVE_TDM
    if (wave == 0) {
        tdm_load_tile(gA, (uint32_t)(uintptr_t)&As[0][0], (uint32_t)K);
        tdm_load_tile(gB, (uint32_t)(uintptr_t)&Bs[0][0], (uint32_t)K);
    }
#else
    {
        uint4 ra0 = *(const uint4*)&gA[(size_t)r0 * K + o0];
        uint4 ra1 = *(const uint4*)&gA[(size_t)r1 * K + o1];
        uint4 rb0 = *(const uint4*)&gB[(size_t)r0 * K + o0];
        uint4 rb1 = *(const uint4*)&gB[(size_t)r1 * K + o1];
        *(uint4*)&As[0][r0 * 80 + o0] = ra0;
        *(uint4*)&As[0][r1 * 80 + o1] = ra1;
        *(uint4*)&Bs[0][r0 * 80 + o0] = rb0;
        *(uint4*)&Bs[0][r1 * 80 + o1] = rb1;
    }
#endif

    int buf = 0;
    for (int kk = 0; kk < K; kk += 64) {
#if HAVE_TDM
        if (wave == 0) __builtin_amdgcn_s_wait_tensorcnt(0);
#endif
        __syncthreads();

        const bool more = (kk + 64) < K;
#if HAVE_TDM
        if (more) {
            if (wave == 0) {
                tdm_load_tile(gA + kk + 64, (uint32_t)(uintptr_t)&As[buf ^ 1][0], (uint32_t)K);
                tdm_load_tile(gB + kk + 64, (uint32_t)(uintptr_t)&Bs[buf ^ 1][0], (uint32_t)K);
            }
            if (kk + 128 < K)
                __builtin_prefetch(&gA[(size_t)(t >> 1) * K + kk + 128], 0, 1);
        }
#else
        uint4 na0, na1, nb0, nb1;
        if (more) {
            na0 = *(const uint4*)&gA[(size_t)r0 * K + kk + 64 + o0];
            na1 = *(const uint4*)&gA[(size_t)r1 * K + kk + 64 + o1];
            nb0 = *(const uint4*)&gB[(size_t)r0 * K + kk + 64 + o0];
            nb1 = *(const uint4*)&gB[(size_t)r1 * K + kk + 64 + o1];
        }
#endif

        v8i af[2], bf[4];
        #pragma unroll
        for (int mi = 0; mi < 2; ++mi)
            af[mi] = load_a_frag(&As[buf][(wm * 32 + mi * 16) * 80], lane);
        #pragma unroll
        for (int ni = 0; ni < 4; ++ni)
            bf[ni] = load_b_frag(&Bs[buf][(wn * 64 + ni * 16) * 80], lane);

        #pragma unroll
        for (int mi = 0; mi < 2; ++mi)
            #pragma unroll
            for (int ni = 0; ni < 4; ++ni)
                acc[mi][ni] = __builtin_amdgcn_wmma_i32_16x16x64_iu8(
                    true, af[mi], true, bf[ni], acc[mi][ni], false, false);

#if !HAVE_TDM
        if (more) {
            *(uint4*)&As[buf ^ 1][r0 * 80 + o0] = na0;
            *(uint4*)&As[buf ^ 1][r1 * 80 + o1] = na1;
            *(uint4*)&Bs[buf ^ 1][r0 * 80 + o0] = nb0;
            *(uint4*)&Bs[buf ^ 1][r1 * 80 + o1] = nb1;
        }
#endif
        buf ^= 1;
    }

    const float sw = wScale[e];
    const int rowSel = (lane >> 4) << 3;
    const int colSel = lane & 15;
    float sMul[2][8];
    #pragma unroll
    for (int mi = 0; mi < 2; ++mi)
        #pragma unroll
        for (int r = 0; r < 8; ++r) {
            int m = mBase + wm * 32 + mi * 16 + rowSel + r;
            sMul[mi][r] = sw * aScale[e * CAP + m];
        }
    #pragma unroll
    for (int mi = 0; mi < 2; ++mi)
        #pragma unroll
        for (int ni = 0; ni < 4; ++ni) {
            int n = nBase + wn * 64 + ni * 16 + colSel;
            #pragma unroll
            for (int r = 0; r < 8; ++r) {
                int m = mBase + wm * 32 + mi * 16 + rowSel + r;
                float v = (float)acc[mi][ni][r] * sMul[mi][r];
                if (GELU) v = gelu_tanh(v);
                Out[((size_t)e * CAP + m) * NN + n] = v;
            }
        }
}


__global__ void k_gather_add(const float* __restrict__ eo, const int* __restrict__ te,
                             const int* __restrict__ tl, const float* __restrict__ tw,
                             float* __restrict__ out) {
    int n = blockIdx.x;
    int t = threadIdx.x;
    int e = te[n], l = tl[n];
    float w = tw[n];
    const float4* src = (const float4*)(eo + ((size_t)e * CAP + l) * DM);
    float4* dst = (float4*)(out + (size_t)n * DM);
    float4 s = src[t];
    float4 d = dst[t];
    d.x += s.x * w; d.y += s.y * w; d.z += s.z * w; d.w += s.w * w;
    dst[t] = d;
}

__global__ void k_aux(const float* __restrict__ fcnt, const float* __restrict__ psum,
                      float* __restrict__ out) {
    if (threadIdx.x == 0) {
        float a = 0.0f;
        #pragma unroll
        for (int e = 0; e < NEXP; ++e)
            a += (fcnt[e] / (float)(N_TOK * 2)) * (psum[e] / (float)N_TOK);
        out[0] = (float)NEXP * a;
    }
}


extern "C" void kernel_launch(void* const* d_in, const int* in_sizes, int n_in,
                              void* d_out, int out_size, void* d_ws, size_t ws_size,
                              hipStream_t stream) {
    (void)in_sizes; (void)n_in; (void)out_size; (void)ws_size;
    const float* x  = (const float*)d_in[0];
    const float* rw = (const float*)d_in[1];
    const float* w1 = (const float*)d_in[2];
    const float* w2 = (const float*)d_in[3];
    float* out = (float*)d_out;

    char* ws = (char*)d_ws;
    size_t off = 0;
    auto seg = [&](size_t bytes) -> char* {
        char* p = ws + off;
        off = (off + bytes + 255) & ~(size_t)255;
        return p;
    };
    int8_t* W1Q   = (int8_t*)seg((size_t)NEXP * DF * DM);
    int8_t* W2Q   = (int8_t*)seg((size_t)NEXP * DM * DF);
    float* WSUM   = (float*)seg(32 * 4);
    float* WSCALE = (float*)seg(32 * 4);
    float* FCNT   = (float*)seg(16 * 4);
    float* PSUM   = (float*)seg(16 * 4);
    int*   TOKE   = (int*)seg(2 * N_TOK * 4);
    float* TOKW   = (float*)seg(2 * N_TOK * 4);
    int*   TOKL   = (int*)seg(N_TOK * 4);
    int*   CCNT   = (int*)seg((N_TOK / 256) * NEXP * 4);
    int*   CBASE  = (int*)seg((N_TOK / 256) * NEXP * 4);
    int8_t* BUFQ  = (int8_t*)seg((size_t)NROWS * DM);
    float* ASCALE = (float*)seg((size_t)NROWS * 4);
    float* H      = (float*)seg((size_t)NROWS * DF * 4);
    int8_t* HQ    = (int8_t*)seg((size_t)NROWS * DF);
    float* HSCALE = (float*)seg((size_t)NROWS * 4);
    float* EO     = (float*)seg((size_t)NROWS * DM * 4);

    k_zero_small<<<1, 64, 0, stream>>>(WSUM, FCNT, PSUM);
    k_wabs_sum<<<dim3(256, NEXP), 256, 0, stream>>>(w1, WSUM);
    k_wabs_sum<<<dim3(256, NEXP), 256, 0, stream>>>(w2, WSUM + 16);
    k_wscale<<<1, 32, 0, stream>>>(WSUM, WSCALE, 1.0f / ((float)DM * (float)DF));
    {
        size_t total = (size_t)NEXP * DM * DF;
        int blocks = (int)((total + 255) / 256);
        k_quant_w_t<<<blocks, 256, 0, stream>>>(w1, WSCALE,      W1Q, DM, DF);
        k_quant_w_t<<<blocks, 256, 0, stream>>>(w2, WSCALE + 16, W2Q, DF, DM);
    }

    k_router<<<N_TOK / 8, 256, 0, stream>>>(x, rw, TOKE, TOKW, FCNT, PSUM);
    k_zero_f32<<<(N_TOK * DM + 1 + 255) / 256, 256, 0, stream>>>(out, N_TOK * DM + 1);

    for (int k = 0; k < 2; ++k) {
        const int* te = TOKE + k * N_TOK;
        const float* tw = TOKW + k * N_TOK;
        k_count_chunks<<<N_TOK / 256, 256, 0, stream>>>(te, CCNT);
        k_scan_chunks<<<1, 32, 0, stream>>>(CCNT, CBASE);
        k_rank_local<<<N_TOK / 256, 256, 0, stream>>>(te, CBASE, TOKL);
        k_zero_u32<<<((NROWS * DM / 4) + 255) / 256, 256, 0, stream>>>((uint32_t*)BUFQ, NROWS * DM / 4);
        k_init_ascale<<<(NROWS + 255) / 256, 256, 0, stream>>>(ASCALE, NROWS);
        k_quant_scatter<<<N_TOK / 8, 256, 0, stream>>>(x, te, TOKL, BUFQ, ASCALE);

        k_moe_gemm_i8<true><<<dim3(DF / 128, CAP / 128, NEXP), 256, 0, stream>>>(
            BUFQ, ASCALE, W1Q, WSCALE, H, DF, DM);
        k_row_quant<<<NROWS / 8, 256, 0, stream>>>(H, DF, HQ, HSCALE);
        k_moe_gemm_i8<false><<<dim3(DM / 128, CAP / 128, NEXP), 256, 0, stream>>>(
            HQ, HSCALE, W2Q, WSCALE + 16, EO, DM, DF);

        k_gather_add<<<N_TOK, 256, 0, stream>>>(EO, te, TOKL, tw, out);
    }

    k_aux<<<1, 32, 0, stream>>>(FCNT, PSUM, out + (size_t)N_TOK * DM);
}
